// GatedDeltaProduct_3444563771922
// MI455X (gfx1250) — matrix operand order repaired and checked by reading, not compiled, not run on hardware
//
#include <hip/hip_runtime.h>

// ---------------------------------------------------------------- constants
static constexpr int B_   = 2;
static constexpr int L_   = 512;
static constexpr int BL_  = B_ * L_;          // 1024
static constexpr int HID  = 2048;
static constexpr int H_   = 8;
static constexpr int DH   = 128;
static constexpr int KD   = H_ * DH;          // 1024
static constexpr int KD2  = KD * 2;           // 2048

// ---------------------------------------------------------------- types
typedef __bf16 v8bf  __attribute__((ext_vector_type(8)));
typedef __bf16 v16bf __attribute__((ext_vector_type(16)));
typedef float  v8f   __attribute__((ext_vector_type(8)));
typedef int          i32x4 __attribute__((ext_vector_type(4)));
typedef int          i32x8 __attribute__((ext_vector_type(8)));
typedef unsigned int u32x4 __attribute__((ext_vector_type(4)));
// address-space qualified element types for the async-to-LDS builtin
typedef __attribute__((address_space(1))) i32x4 gas_i32x4;   // global
typedef __attribute__((address_space(3))) i32x4 las_i32x4;   // LDS

// ------------------------------------------------- gfx1250 feature probes
#if defined(__gfx1250__)
#if __has_builtin(__builtin_amdgcn_global_load_async_to_lds_b128)
#define HAVE_ASYNC 1
#endif
#if __has_builtin(__builtin_amdgcn_tensor_load_to_lds)
#define HAVE_TDM 1
#endif
#endif

#define TM 128
#define TN 128
#define TK 32

// =========================================================================
// Tile staging helpers
// =========================================================================
// Per-lane 16B copies of a 128x32 bf16 tile (rows stride = K elements).
__device__ __forceinline__ void stage_tile_async(const __bf16* __restrict__ gsrc,
                                                 int row_stride, __bf16 (*lds)[TK],
                                                 int tid)
{
#pragma unroll
    for (int c = tid; c < 512; c += 256) {
        const int row = c >> 2;
        const int seg = c & 3;
        const __bf16* g = gsrc + (size_t)row * row_stride + seg * 8;
        __bf16* l = &lds[row][seg * 8];
#if HAVE_ASYNC
        __builtin_amdgcn_global_load_async_to_lds_b128(
            (gas_i32x4*)g, (las_i32x4*)l, 0, 0);
#else
        *(uint4*)l = *(const uint4*)g;
#endif
    }
}

#if HAVE_TDM
// TDM: one 2-D tile descriptor moves the whole 128x32 bf16 tile to LDS.
__device__ __forceinline__ void stage_tile_tdm(const __bf16* gsrc, int K, void* ldsp)
{
    const unsigned long long ga = (unsigned long long)(uintptr_t)gsrc;
    const unsigned int lds_addr = (unsigned int)(uintptr_t)ldsp;
    u32x4 g0;
    g0[0] = 1u;                                               // count=1, user mode
    g0[1] = lds_addr;                                         // lds_addr [63:32]
    g0[2] = (unsigned int)ga;                                 // global_addr lo
    g0[3] = ((unsigned int)(ga >> 32) & 0x01FFFFFFu)          // global_addr [56:32]
            | (2u << 30);                                     // type=2 ("image")
    i32x8 g1;
    g1[0] = 0x00010000;                                       // data_size=1 (2 bytes)
    g1[1] = (int)(((unsigned)K & 0xFFFFu) << 16);             // tensor_dim0 lo16
    g1[2] = (int)((((unsigned)K >> 16) & 0xFFFFu)             // tensor_dim0 hi16
                  | (128u << 16));                            // tensor_dim1 lo16
    g1[3] = (int)(32u << 16);                                 // tile_dim0 = 32
    g1[4] = 128;                                              // tile_dim1 = 128
    g1[5] = K;                                                // tensor_dim0_stride lo32
    g1[6] = 0;
    g1[7] = 0;
    const i32x4 gz4 = {0, 0, 0, 0};
    const i32x8 gz8 = {0, 0, 0, 0, 0, 0, 0, 0};
    __builtin_amdgcn_tensor_load_to_lds(g0, g1, gz4, gz4, gz8, 0);
}
#endif

__device__ __forceinline__ void wait_stage_and_sync()
{
#if HAVE_ASYNC
#if __has_builtin(__builtin_amdgcn_s_wait_asynccnt)
    __builtin_amdgcn_s_wait_asynccnt(0);
#else
    asm volatile("s_wait_asynccnt 0x0" ::: "memory");
#endif
#endif
#if HAVE_TDM
    __builtin_amdgcn_s_wait_tensorcnt((short)0);
#endif
    __syncthreads();
}

// =========================================================================
// bf16 WMMA GEMM:  C(MxN,f32) = A(MxK,bf16 row-major) * Bt(NxK,bf16)^T
// 128x128 block tile, 8 waves x (2x4 of 16x16x32), double-buffered LDS,
// A tile via TDM, B tile via async-to-LDS.
// =========================================================================
__global__ __launch_bounds__(256)
void gemm_bf16_wmma(const __bf16* __restrict__ A, const __bf16* __restrict__ Bt,
                    float* __restrict__ C, int M, int N, int K)
{
    (void)M;
    __shared__ __align__(128) __bf16 As[2][TM][TK];   // 2 x 8KB
    __shared__ __align__(128) __bf16 Bs[2][TN][TK];   // 2 x 8KB

    const int tid  = threadIdx.x;
    const int wave = tid >> 5;
    const int lane = tid & 31;
    const int half = lane >> 4;
    const int l16  = lane & 15;

    const int mblk = blockIdx.y * TM;
    const int nblk = blockIdx.x * TN;
    const int wrow = wave & 3;    // 4 x 32-row subtiles
    const int wcol = wave >> 2;   // 2 x 64-col subtiles

    const v8f vzero = {0.f,0.f,0.f,0.f,0.f,0.f,0.f,0.f};
    v8f acc[2][4];
#pragma unroll
    for (int mi = 0; mi < 2; ++mi)
#pragma unroll
        for (int ni = 0; ni < 4; ++ni) acc[mi][ni] = vzero;

    const __bf16* Abase = A  + (size_t)mblk * K;
    const __bf16* Bbase = Bt + (size_t)nblk * K;

    // ---- preload tile 0 ------------------------------------------------
#if HAVE_TDM
    if (tid == 0) stage_tile_tdm(Abase, K, &As[0][0][0]);
#else
    stage_tile_async(Abase, K, As[0], tid);
#endif
    stage_tile_async(Bbase, K, Bs[0], tid);
    wait_stage_and_sync();

    const int nsteps = K / TK;
    int p = 0;
    for (int s = 0; s < nsteps; ++s) {
        // ---- kick off next tile DMA while we compute -------------------
        if (s + 1 < nsteps) {
            const int kn = (s + 1) * TK;
#if HAVE_TDM
            if (tid == 0) stage_tile_tdm(Abase + kn, K, &As[p ^ 1][0][0]);
#else
            stage_tile_async(Abase + kn, K, As[p ^ 1], tid);
#endif
            stage_tile_async(Bbase + kn, K, Bs[p ^ 1], tid);
        }

        // ---- A/B fragments in contiguous k order (k = 16*half + s) ------------------
        v16bf afrag[2];
#pragma unroll
        for (int mi = 0; mi < 2; ++mi) {
            const int row = wrow * 32 + mi * 16 + l16;
            v8bf lo = *(const v8bf*)&As[p][row][half * 16];      // K = h*16..h*16+7
            v8bf hi = *(const v8bf*)&As[p][row][half * 16 + 8];  // K = h*16+8..h*16+15
            v16bf a;
#pragma unroll
            for (int i = 0; i < 8; ++i) { a[i] = lo[i]; a[i + 8] = hi[i]; }
            afrag[mi] = a;
        }
        v16bf bfrag[4];
#pragma unroll
        for (int ni = 0; ni < 4; ++ni) {
            const int col = wcol * 64 + ni * 16 + l16;
            bfrag[ni] = *(const v16bf*)&Bs[p][col][half * 16];   // K = h*16..h*16+15
        }

        // ---- 8 WMMAs per K-step ----------------------------------------
#pragma unroll
        for (int mi = 0; mi < 2; ++mi)
#pragma unroll
            for (int ni = 0; ni < 4; ++ni)
                acc[mi][ni] = __builtin_amdgcn_wmma_f32_16x16x32_bf16(
                    false, afrag[mi], false, bfrag[ni],
                    (short)0, acc[mi][ni], false, false);

        wait_stage_and_sync();
        p ^= 1;
    }

    // ---- store C per 16x16 f32 C/D layout (VGPR r -> M = half*8 + r) ---
#pragma unroll
    for (int mi = 0; mi < 2; ++mi)
#pragma unroll
        for (int ni = 0; ni < 4; ++ni) {
            const int mbase = mblk + wrow * 32 + mi * 16 + half * 8;
            const int ncol  = nblk + wcol * 64 + ni * 16 + l16;
#pragma unroll
            for (int r = 0; r < 8; ++r)
                C[(size_t)(mbase + r) * N + ncol] = acc[mi][ni][r];
        }
}

// =========================================================================
// pre-pass: f32 -> bf16 cast
// =========================================================================
__global__ __launch_bounds__(256)
void cast_bf16(const float* __restrict__ in, __bf16* __restrict__ out, int n)
{
    const int i = (blockIdx.x * 256 + threadIdx.x) * 4;
    if (i < n) {
        float4 f = *(const float4*)(in + i);
        out[i + 0] = (__bf16)f.x;
        out[i + 1] = (__bf16)f.y;
        out[i + 2] = (__bf16)f.z;
        out[i + 3] = (__bf16)f.w;
    }
}

// =========================================================================
// pre-pass: W(KxN,f32) -> Wt(NxK,bf16) transpose + cast, 32x32 LDS tiles
// =========================================================================
__global__ __launch_bounds__(256)
void transpose_cast_bf16(const float* __restrict__ W, __bf16* __restrict__ Wt,
                         int K, int N)
{
    __shared__ float t[32][33];
    const int nb = blockIdx.x * 32;
    const int kb = blockIdx.y * 32;
    const int tx = threadIdx.x & 31;
    const int ty = threadIdx.x >> 5;      // 8 row-groups
#pragma unroll
    for (int r = ty; r < 32; r += 8)
        t[r][tx] = W[(size_t)(kb + r) * N + nb + tx];
    __syncthreads();
#pragma unroll
    for (int r = ty; r < 32; r += 8)
        Wt[(size_t)(nb + r) * K + kb + tx] = (__bf16)t[tx][r];
}

// =========================================================================
// beta = 2*sigmoid(x@Wb), g = -exp(A_log)*softplus(x@Wa+dt_bias)
// =========================================================================
__global__ __launch_bounds__(256)
void proj_small(const float* __restrict__ x, const float* __restrict__ Wb,
                const float* __restrict__ Wa, const float* __restrict__ A_log,
                const float* __restrict__ dtb,
                float* __restrict__ beta, float* __restrict__ g)
{
    const int bl   = blockIdx.x;
    const int wave = threadIdx.x >> 5;
    const int lane = threadIdx.x & 31;
    const float* xr = x + (size_t)bl * HID;

    for (int o = wave; o < 24; o += 8) {
        const float* w  = (o < 16) ? (Wb + o) : (Wa + (o - 16));
        const int    ld = (o < 16) ? 16 : 8;
        float acc = 0.f;
        for (int k = lane; k < HID; k += 32)
            acc += xr[k] * w[(size_t)k * ld];
#pragma unroll
        for (int off = 16; off; off >>= 1)
            acc += __shfl_xor(acc, off, 32);
        if (lane == 0) {
            if (o < 16) {
                beta[bl * 16 + o] = 2.f / (1.f + expf(-acc));
            } else {
                const int h = o - 16;
                const float z  = acc + dtb[h];
                const float sp = (z > 20.f) ? z : log1pf(expf(z));
                g[bl * 8 + h] = -expf(A_log[h]) * sp;
            }
        }
    }
}

// =========================================================================
// causal depthwise conv(4) + SiLU + optional per-head L2 norm
// =========================================================================
__global__ __launch_bounds__(128)
void conv_silu_norm(const float* __restrict__ in, const float* __restrict__ w,
                    float* __restrict__ out, int C, int mode, float scale)
{
    const int chunks = C >> 7;
    const int chunk  = blockIdx.x % chunks;
    const int bl     = blockIdx.x / chunks;
    const int b = bl / L_, l = bl % L_;
    const int c = chunk * 128 + threadIdx.x;

    float acc = 0.f;
#pragma unroll
    for (int t = 0; t < 4; ++t) {
        const int ll = l - 3 + t;
        if (ll >= 0) acc += w[c * 4 + t] * in[((size_t)(b * L_ + ll)) * C + c];
    }
    float y = acc / (1.f + expf(-acc));   // SiLU

    if (mode) {
        __shared__ float red[128];
        red[threadIdx.x] = y * y;
        __syncthreads();
        for (int s = 64; s; s >>= 1) {
            if (threadIdx.x < s) red[threadIdx.x] += red[threadIdx.x + s];
            __syncthreads();
        }
        y *= scale / fmaxf(sqrtf(red[0]), 1e-12f);
    }
    out[(size_t)bl * C + c] = y;
}

// =========================================================================
// Sequential gated delta scan; lane owns one state column (128 f32 VGPRs)
// =========================================================================
__global__ __launch_bounds__(128)
void scan_kernel(const float* __restrict__ qc, const float* __restrict__ kc,
                 const float* __restrict__ vc, const float* __restrict__ beta,
                 const float* __restrict__ gg,
                 float* __restrict__ ob, float* __restrict__ state_out)
{
    const int b = blockIdx.x >> 3;
    const int h = blockIdx.x & 7;
    const int v = threadIdx.x;

    __shared__ float qs[128];
    __shared__ float ks[2][128];

    float st[128];
#pragma unroll
    for (int k = 0; k < 128; ++k) st[k] = 0.f;

    for (int l = 0; l < L_; ++l) {
        const int bl = b * L_ + l;
        qs[v]    = qc[(size_t)bl * KD  + h * DH + v];
        ks[0][v] = kc[(size_t)bl * KD2 + h * DH + v];
        ks[1][v] = kc[(size_t)bl * KD2 + KD + h * DH + v];
        const float vv0 = vc[(size_t)bl * KD2 + h * DH + v];
        const float vv1 = vc[(size_t)bl * KD2 + KD + h * DH + v];
        const float b0  = beta[bl * 16 + h];
        const float b1  = beta[bl * 16 + 8 + h];
        const float eg  = expf(gg[bl * 8 + h]);
        __syncthreads();

        float corr0 = 0.f;
#pragma unroll
        for (int k = 0; k < 128; ++k) {
            const float s = st[k] * eg;
            st[k] = s;
            corr0 += ks[0][k] * s;
        }
        const float u0 = (vv0 - corr0) * b0;

        float corr1 = 0.f;
#pragma unroll
        for (int k = 0; k < 128; ++k) {
            const float s = st[k] + ks[0][k] * u0;
            st[k] = s;
            corr1 += ks[1][k] * s;
        }
        const float u1 = (vv1 - corr1) * b1;

        float o = 0.f;
#pragma unroll
        for (int k = 0; k < 128; ++k) {
            const float s = st[k] + ks[1][k] * u1;
            st[k] = s;
            o += qs[k] * s;
        }
        ob[(size_t)bl * KD + h * DH + v] = o;
        __syncthreads();
    }

#pragma unroll
    for (int k = 0; k < 128; ++k)
        state_out[((size_t)(b * H_ + h) * DH + k) * DH + v] = st[k];
}

// =========================================================================
// RMSNorm + norm_w * silu(gate); writes bf16 for the final WMMA GEMM
// =========================================================================
__global__ __launch_bounds__(128)
void norm_gate(const float* __restrict__ ob, const float* __restrict__ gp,
               const float* __restrict__ nw, __bf16* __restrict__ onb)
{
    const int h  = blockIdx.x & 7;
    const int bl = blockIdx.x >> 3;
    const int v  = threadIdx.x;

    const float o = ob[(size_t)bl * KD + h * DH + v];
    __shared__ float red[128];
    red[v] = o * o;
    __syncthreads();
    for (int s = 64; s; s >>= 1) {
        if (v < s) red[v] += red[v + s];
        __syncthreads();
    }
    const float r    = rsqrtf(red[0] / (float)DH + 1e-5f);
    const float gate = gp[(size_t)bl * KD + h * DH + v];
    const float sg   = gate / (1.f + expf(-gate));
    onb[(size_t)bl * KD + h * DH + v] = (__bf16)(o * r * nw[v] * sg);
}

// =========================================================================
extern "C" void kernel_launch(void* const* d_in, const int* in_sizes, int n_in,
                              void* d_out, int out_size, void* d_ws, size_t ws_size,
                              hipStream_t stream)
{
    (void)in_sizes; (void)n_in; (void)out_size; (void)ws_size;

    const float* x     = (const float*)d_in[0];
    const float* Wq    = (const float*)d_in[1];
    const float* Wk    = (const float*)d_in[2];
    const float* Wv    = (const float*)d_in[3];
    const float* Wb    = (const float*)d_in[4];
    const float* Wa    = (const float*)d_in[5];
    const float* A_log = (const float*)d_in[6];
    const float* dtb   = (const float*)d_in[7];
    const float* cqw   = (const float*)d_in[8];
    const float* ckw   = (const float*)d_in[9];
    const float* cvw   = (const float*)d_in[10];
    const float* Wg    = (const float*)d_in[11];
    const float* nw    = (const float*)d_in[12];
    const float* Wo    = (const float*)d_in[13];

    float* out       = (float*)d_out;                 // (B,L,HID)
    float* state_out = out + (size_t)BL_ * HID;       // (B,H,128,128)

    // f32 scratch
    float* ws   = (float*)d_ws;
    float* qp   = ws;   ws += (size_t)BL_ * KD;
    float* kp   = ws;   ws += (size_t)BL_ * KD2;
    float* vp   = ws;   ws += (size_t)BL_ * KD2;
    float* gp   = ws;   ws += (size_t)BL_ * KD;
    float* qc   = ws;   ws += (size_t)BL_ * KD;
    float* kc   = ws;   ws += (size_t)BL_ * KD2;
    float* vc   = ws;   ws += (size_t)BL_ * KD2;
    float* beta = ws;   ws += (size_t)BL_ * 16;
    float* gg   = ws;   ws += (size_t)BL_ * 8;
    float* ob   = ws;   ws += (size_t)BL_ * KD;
    // bf16 scratch (packed operands)
    __bf16* wsb = (__bf16*)ws;
    __bf16* xb  = wsb;  wsb += (size_t)BL_ * HID;     // x cast
    __bf16* Wqt = wsb;  wsb += (size_t)KD  * HID;     // W^T casts
    __bf16* Wkt = wsb;  wsb += (size_t)KD2 * HID;
    __bf16* Wvt = wsb;  wsb += (size_t)KD2 * HID;
    __bf16* Wgt = wsb;  wsb += (size_t)KD  * HID;
    __bf16* Wot = wsb;  wsb += (size_t)HID * KD;
    __bf16* onb = wsb;  wsb += (size_t)BL_ * KD;

    const dim3 blk256(256);

    // ---- operand packing (one cheap bandwidth pass each) ---------------
    cast_bf16<<<dim3((BL_ * HID / 4) / 256), blk256, 0, stream>>>(x, xb, BL_ * HID);
    transpose_cast_bf16<<<dim3(KD  / 32, HID / 32), blk256, 0, stream>>>(Wq, Wqt, HID, KD);
    transpose_cast_bf16<<<dim3(KD2 / 32, HID / 32), blk256, 0, stream>>>(Wk, Wkt, HID, KD2);
    transpose_cast_bf16<<<dim3(KD2 / 32, HID / 32), blk256, 0, stream>>>(Wv, Wvt, HID, KD2);
    transpose_cast_bf16<<<dim3(KD  / 32, HID / 32), blk256, 0, stream>>>(Wg, Wgt, HID, KD);
    transpose_cast_bf16<<<dim3(HID / 32, KD  / 32), blk256, 0, stream>>>(Wo, Wot, KD, HID);

    // ---- big projections via WMMA GEMM (TDM/async staged) --------------
    gemm_bf16_wmma<<<dim3(KD  / TN, BL_ / TM), blk256, 0, stream>>>(xb, Wqt, qp, BL_, KD,  HID);
    gemm_bf16_wmma<<<dim3(KD2 / TN, BL_ / TM), blk256, 0, stream>>>(xb, Wkt, kp, BL_, KD2, HID);
    gemm_bf16_wmma<<<dim3(KD2 / TN, BL_ / TM), blk256, 0, stream>>>(xb, Wvt, vp, BL_, KD2, HID);
    gemm_bf16_wmma<<<dim3(KD  / TN, BL_ / TM), blk256, 0, stream>>>(xb, Wgt, gp, BL_, KD,  HID);

    // ---- skinny beta / g projections ------------------------------------
    proj_small<<<dim3(BL_), blk256, 0, stream>>>(x, Wb, Wa, A_log, dtb, beta, gg);

    // ---- depthwise conv + SiLU (+ per-head L2 norm for q,k) -------------
    const float inv_sqrt_dh = 0.0883883476483184f;    // 1/sqrt(128)
    conv_silu_norm<<<dim3(BL_ * (KD  / 128)), dim3(128), 0, stream>>>(qp, cqw, qc, KD,  1, inv_sqrt_dh);
    conv_silu_norm<<<dim3(BL_ * (KD2 / 128)), dim3(128), 0, stream>>>(kp, ckw, kc, KD2, 1, 1.0f);
    conv_silu_norm<<<dim3(BL_ * (KD2 / 128)), dim3(128), 0, stream>>>(vp, cvw, vc, KD2, 0, 1.0f);

    // ---- sequential gated delta-product scan ----------------------------
    scan_kernel<<<dim3(B_ * H_), dim3(128), 0, stream>>>(qc, kc, vc, beta, gg, ob, state_out);

    // ---- RMSNorm + gate (emits bf16) ------------------------------------
    norm_gate<<<dim3(BL_ * H_), dim3(128), 0, stream>>>(ob, gp, nw, onb);

    // ---- final output projection via WMMA GEMM --------------------------
    gemm_bf16_wmma<<<dim3(HID / TN, BL_ / TM), blk256, 0, stream>>>(onb, Wot, out, BL_, HID, KD);
}
